// Block_2551210574650
// MI455X (gfx1250) — matrix operand order repaired and checked by reading, not compiled, not run on hardware
//
#include <hip/hip_runtime.h>

// ---------------------------------------------------------------------------
// Problem constants (match the reference)
// ---------------------------------------------------------------------------
constexpr int Bv  = 4;
constexpr int Tv  = 2048;
constexpr int Cv  = 768;
constexpr int Hv  = 12;
constexpr int Ev  = 8;
constexpr int DFv = 4 * Cv;      // 3072
constexpr int Nv  = Bv * Tv;     // 8192 tokens
constexpr int Dv  = Cv / Hv;     // 64 head dim

typedef __attribute__((ext_vector_type(4)))  int    v4i;
typedef __attribute__((ext_vector_type(8)))  __bf16 v8bf;
typedef __attribute__((ext_vector_type(16))) __bf16 v16bf;
typedef __attribute__((ext_vector_type(8)))  float  v8f;

#define GLOBAL_AS __attribute__((address_space(1)))
#define LDS_AS    __attribute__((address_space(3)))

#if defined(__gfx1250__) && __has_builtin(__builtin_amdgcn_global_load_async_to_lds_b128)
#define HAVE_ASYNC_LDS 1
#else
#define HAVE_ASYNC_LDS 0
#endif

__device__ __forceinline__ void wait_async_zero() {
#if HAVE_ASYNC_LDS
#if __has_builtin(__builtin_amdgcn_s_wait_asynccnt)
    __builtin_amdgcn_s_wait_asynccnt(0);
#else
    asm volatile("s_wait_asynccnt 0x0" ::: "memory");
#endif
#endif
}

__device__ __forceinline__ v16bf vcat(v8bf lo, v8bf hi) {
    v16bf r;
#pragma unroll
    for (int i = 0; i < 8; i++) { r[i] = lo[i]; r[i + 8] = hi[i]; }
    return r;
}

__device__ __forceinline__ v8f wmma_bf16(v16bf a, v16bf b, v8f c) {
    // D = A(16x32 bf16) * B(32x16 bf16) + C(16x16 f32)
    return __builtin_amdgcn_wmma_f32_16x16x32_bf16(false, a, false, b,
                                                   (short)0, c, false, false);
}

// ---------------------------------------------------------------------------
// fp32 -> bf16 cast (weights / activations), grid-stride
// ---------------------------------------------------------------------------
__global__ void cast_bf16_kernel(const float* __restrict__ in,
                                 __bf16* __restrict__ out, long long n) {
    long long i = (long long)blockIdx.x * blockDim.x + threadIdx.x;
    long long stride = (long long)gridDim.x * blockDim.x;
    for (; i < n; i += stride) out[i] = (__bf16)in[i];
}

// ---------------------------------------------------------------------------
// LayerNorm: one wave per row, bf16 output
// ---------------------------------------------------------------------------
__global__ __launch_bounds__(256)
void ln_kernel(const float* __restrict__ X, const float* __restrict__ g,
               const float* __restrict__ bb, __bf16* __restrict__ out,
               int Nrows, int C) {
    int row  = blockIdx.x * (blockDim.x >> 5) + (threadIdx.x >> 5);
    int lane = threadIdx.x & 31;
    if (row >= Nrows) return;
    const float* x = X + (long long)row * C;
    float s = 0.f, s2 = 0.f;
    for (int c = lane; c < C; c += 32) { float v = x[c]; s += v; s2 += v * v; }
#pragma unroll
    for (int off = 16; off; off >>= 1) {
        s  += __shfl_xor(s,  off, 32);
        s2 += __shfl_xor(s2, off, 32);
    }
    float mean = s / C;
    float var  = s2 / C - mean * mean;
    float rstd = rsqrtf(var + 1e-5f);
    __bf16* o = out + (long long)row * C;
    for (int c = lane; c < C; c += 32)
        o[c] = (__bf16)((x[c] - mean) * rstd * g[c] + bb[c]);
}

// ---------------------------------------------------------------------------
// Generic bf16 WMMA GEMM: Out[M,N] = A[M,K] @ B[K,N] (+bias)(+res)(relu)
// 128x128 block tile, BK=32, 256 threads = 8 waves (4 M x 2 N),
// wave tile 32x64 = 2x4 WMMA tiles. A tile staged with async global->LDS
// copies (ASYNCcnt path); B tile transposed through VGPRs. Optional row
// gather (MoE in) and weighted atomic scatter-add (MoE out).
// ---------------------------------------------------------------------------
template <bool HAS_BIAS, bool HAS_RES, bool RELU, bool OUT_BF16,
          bool GATHER, bool SCATTER>
__global__ __launch_bounds__(256)
void gemm_bf16(const __bf16* __restrict__ A, int lda,
               const __bf16* __restrict__ Bw, long long strideB,
               const float* __restrict__ bias, int strideBias,
               const float* __restrict__ Res,
               void* __restrict__ Out, int ldo,
               int Mrows, int Ncol, int Ktot,
               const int* __restrict__ rowlist,
               const float* __restrict__ wlist,
               const int* __restrict__ segOff) {
    __shared__ __bf16 As[128][48];   // 48 elts = 96 B rows (16B multiple)
    __shared__ __bf16 Bs[128][48];   // stored transposed: Bs[n][k]

    int e = blockIdx.z;
    int rowBeg = 0, rowEnd = Mrows;
    if (segOff) { rowBeg = segOff[e]; rowEnd = segOff[e + 1]; }
    int m0 = rowBeg + blockIdx.y * 128;
    if (m0 >= rowEnd) return;
    int n0 = blockIdx.x * 128;

    const __bf16* Bm = Bw + (long long)e * strideB;
    const float*  bi = HAS_BIAS ? bias + (long long)e * strideBias : nullptr;

    int wave = threadIdx.x >> 5, lane = threadIdx.x & 31;
    int wm = wave & 3, wn = wave >> 2;
    int half = lane >> 4, r = lane & 15;

    v8f acc[2][4] = {};

    for (int k0 = 0; k0 < Ktot; k0 += 32) {
        // --- stage A tile: 256 threads x 16 bf16, async global -> LDS ---
        {
            int rr = threadIdx.x >> 1;
            int cc = (threadIdx.x & 1) * 16;
            int gr = m0 + rr;
            if (gr < rowEnd) {
                int ar = GATHER ? rowlist[gr] : gr;
                const __bf16* p = A + (long long)ar * lda + k0 + cc;
#if HAVE_ASYNC_LDS
                __builtin_amdgcn_global_load_async_to_lds_b128(
                    (GLOBAL_AS v4i*)p,
                    (LDS_AS v4i*)&As[rr][cc], 0, 0);
                __builtin_amdgcn_global_load_async_to_lds_b128(
                    (GLOBAL_AS v4i*)(p + 8),
                    (LDS_AS v4i*)&As[rr][cc + 8], 0, 0);
#else
                v8bf lo = *(const v8bf*)p;
                v8bf hi = *(const v8bf*)(p + 8);
                *(v8bf*)&As[rr][cc]     = lo;
                *(v8bf*)&As[rr][cc + 8] = hi;
#endif
                if (k0 + 32 < Ktot) __builtin_prefetch(p + 32, 0, 1);
            } else {
#if !HAVE_ASYNC_LDS
                v8bf z = {};
                *(v8bf*)&As[rr][cc]     = z;
                *(v8bf*)&As[rr][cc + 8] = z;
#endif
            }
        }
        // --- stage B tile transposed (through VGPRs) ---
        {
            int kk = threadIdx.x >> 3;          // 0..31
            int nb = (threadIdx.x & 7) * 16;    // 0..112
            const __bf16* p = Bm + (long long)(k0 + kk) * Ncol + n0 + nb;
            v8bf lo = *(const v8bf*)p;
            v8bf hi = *(const v8bf*)(p + 8);
            if (k0 + 32 < Ktot) __builtin_prefetch(p + 32LL * Ncol, 0, 1);
#pragma unroll
            for (int i = 0; i < 8; i++) {
                Bs[nb + i][kk]     = lo[i];
                Bs[nb + 8 + i][kk] = hi[i];
            }
        }
        wait_async_zero();                 // ASYNCcnt not covered by barrier
        __syncthreads();

        // --- fragments per documented CDNA5 layouts ---
        v16bf af[2], bfv[4];
#pragma unroll
        for (int mt = 0; mt < 2; mt++) {
            const __bf16* p = &As[wm * 32 + mt * 16 + r][half * 8];
            af[mt] = vcat(*(const v8bf*)p, *(const v8bf*)(p + 16));
        }
#pragma unroll
        for (int nt = 0; nt < 4; nt++) {
            const __bf16* p = &Bs[wn * 64 + nt * 16 + r][half * 8];
            bfv[nt] = vcat(*(const v8bf*)p, *(const v8bf*)(p + 16));
        }
#pragma unroll
        for (int mt = 0; mt < 2; mt++)
#pragma unroll
            for (int nt = 0; nt < 4; nt++)
                acc[mt][nt] = wmma_bf16(af[mt], bfv[nt], acc[mt][nt]);
        __syncthreads();
    }

    // --- epilogue (C layout: row = base + 8*half + j, col = base + r) ---
#pragma unroll
    for (int mt = 0; mt < 2; mt++) {
        int rowb = m0 + wm * 32 + mt * 16 + half * 8;
#pragma unroll
        for (int nt = 0; nt < 4; nt++) {
            int col = n0 + wn * 64 + nt * 16 + r;
#pragma unroll
            for (int j = 0; j < 8; j++) {
                int gr = rowb + j;
                if (gr >= rowEnd) continue;
                float v = acc[mt][nt][j];
                if (HAS_BIAS) v += bi[col];
                if (SCATTER) {
                    int tok  = rowlist[gr];
                    float wk = wlist[gr];
                    atomicAdd((float*)Out + (long long)tok * ldo + col, v * wk);
                } else {
                    if (HAS_RES) v += Res[(long long)gr * ldo + col];
                    if (RELU) v = fmaxf(v, 0.f);
                    if (OUT_BF16)
                        ((__bf16*)Out)[(long long)gr * ldo + col] = (__bf16)v;
                    else
                        ((float*)Out)[(long long)gr * ldo + col] = v;
                }
            }
        }
    }
}

// ---------------------------------------------------------------------------
// split qkv[N,3C] bf16 -> q/k/v [B,H,T,D] bf16
// ---------------------------------------------------------------------------
__global__ void split_qkv_kernel(const __bf16* __restrict__ qkv,
                                 __bf16* __restrict__ q, __bf16* __restrict__ k,
                                 __bf16* __restrict__ v) {
    long long total = (long long)Nv * Cv;
    long long i = (long long)blockIdx.x * blockDim.x + threadIdx.x;
    long long stride = (long long)gridDim.x * blockDim.x;
    for (; i < total; i += stride) {
        int d = (int)(i % Dv);
        int h = (int)((i / Dv) % Hv);
        int t = (int)((i / ((long long)Dv * Hv)) % Tv);
        int b = (int)(i / ((long long)Dv * Hv * Tv));
        long long row = (long long)b * Tv + t;
        long long dst = (((long long)(b * Hv + h)) * Tv + t) * Dv + d;
        q[dst] = qkv[row * (3 * Cv) + h * Dv + d];
        k[dst] = qkv[row * (3 * Cv) + Cv + h * Dv + d];
        v[dst] = qkv[row * (3 * Cv) + 2 * Cv + h * Dv + d];
    }
}

// ---------------------------------------------------------------------------
// Flash attention: grid (T/64, H, B), 128 threads = 4 waves; each wave owns a
// 16-row q tile, streams s in chunks of 32, WMMA for QK^T and PV, online
// softmax with stats in LDS, V chunk transposed into LDS cooperatively.
// Y written as [B,T,C] bf16 (head-concat) for the proj GEMM.
// ---------------------------------------------------------------------------
__global__ __launch_bounds__(128)
void attn_kernel(const __bf16* __restrict__ Q, const __bf16* __restrict__ Kk,
                 const __bf16* __restrict__ V, __bf16* __restrict__ Y) {
    __shared__ __bf16 Ps[4][16][32];
    __shared__ float  mrow[4][16];
    __shared__ float  lrow[4][16];
    __shared__ __bf16 Vts[64][32];     // V chunk transposed: Vts[d][s]

    int wave = threadIdx.x >> 5, lane = threadIdx.x & 31;
    int half = lane >> 4, r = lane & 15;
    int b = blockIdx.z, h = blockIdx.y;
    int q0 = blockIdx.x * 64;
    int qb = q0 + wave * 16;
    long long base = ((long long)(b * Hv + h)) * Tv * Dv;

    // q fragments: 16x64 as two K=32 A-frags
    v16bf qa[2];
    {
        const __bf16* p = Q + base + (long long)(qb + r) * Dv;
#pragma unroll
        for (int kk = 0; kk < 2; kk++) {
            v8bf lo = *(const v8bf*)(p + kk * 32 + half * 8);
            v8bf hi = *(const v8bf*)(p + kk * 32 + half * 8 + 16);
            qa[kk] = vcat(lo, hi);
        }
    }
    v8f o[4] = {};
    if (lane < 16) { mrow[wave][lane] = -1e30f; lrow[wave][lane] = 0.f; }

    const float scale = 0.125f;            // 1/sqrt(64)
    int nchunk = (q0 + 64) / 32;           // causal bound for this block
    for (int ch = 0; ch < nchunk; ch++) {
        int s0 = ch * 32;
        __syncthreads();                   // Vts reuse + stats visibility
        // stage V chunk transposed (block cooperative)
        {
            int sl = threadIdx.x & 31;
            int dg = threadIdx.x >> 5;     // 4 groups of 16 d
            const __bf16* p = V + base + (long long)(s0 + sl) * Dv + dg * 16;
            v8bf lo = *(const v8bf*)p;
            v8bf hi = *(const v8bf*)(p + 8);
#pragma unroll
            for (int i = 0; i < 8; i++) {
                Vts[dg * 16 + i][sl]     = lo[i];
                Vts[dg * 16 + 8 + i][sl] = hi[i];
            }
        }
        // S = q @ k^T for 32 columns (two 16x16 tiles)
        v8f S[2];
#pragma unroll
        for (int g = 0; g < 2; g++) {
            int s = s0 + g * 16 + r;
            v8f c = {};
#pragma unroll
            for (int kk = 0; kk < 2; kk++) {
                const __bf16* p = Kk + base + (long long)s * Dv + kk * 32 + half * 8;
                v16bf kf = vcat(*(const v8bf*)p, *(const v8bf*)(p + 16));
                c = wmma_bf16(qa[kk], kf, c);
            }
            S[g] = c;
        }
        // scale + causal mask
#pragma unroll
        for (int g = 0; g < 2; g++) {
            int col = s0 + g * 16 + r;
#pragma unroll
            for (int j = 0; j < 8; j++) {
                int row = qb + half * 8 + j;
                float v = S[g][j] * scale;
                S[g][j] = (col <= row) ? v : -1e30f;
            }
        }
        // online softmax: row reductions across the 16 lanes of each half
        float mnew[8], alpha[8], rsum[8];
#pragma unroll
        for (int j = 0; j < 8; j++) {
            float mv = fmaxf(S[0][j], S[1][j]);
#pragma unroll
            for (int off = 1; off < 16; off <<= 1)
                mv = fmaxf(mv, __shfl_xor(mv, off, 32));
            float mo = mrow[wave][half * 8 + j];
            mnew[j]  = fmaxf(mo, mv);
            alpha[j] = __expf(mo - mnew[j]);
        }
#pragma unroll
        for (int g = 0; g < 2; g++)
#pragma unroll
            for (int j = 0; j < 8; j++)
                S[g][j] = __expf(S[g][j] - mnew[j]);
#pragma unroll
        for (int j = 0; j < 8; j++) {
            float sv = S[0][j] + S[1][j];
#pragma unroll
            for (int off = 1; off < 16; off <<= 1)
                sv += __shfl_xor(sv, off, 32);
            rsum[j] = sv;
        }
        if (r == 0) {
#pragma unroll
            for (int j = 0; j < 8; j++) {
                mrow[wave][half * 8 + j] = mnew[j];
                lrow[wave][half * 8 + j] =
                    lrow[wave][half * 8 + j] * alpha[j] + rsum[j];
            }
        }
#pragma unroll
        for (int dc = 0; dc < 4; dc++)
#pragma unroll
            for (int j = 0; j < 8; j++) o[dc][j] *= alpha[j];
        // P -> LDS (bf16) in row-major for A-frag reload
#pragma unroll
        for (int g = 0; g < 2; g++)
#pragma unroll
            for (int j = 0; j < 8; j++)
                Ps[wave][half * 8 + j][g * 16 + r] = (__bf16)S[g][j];
        __syncthreads();
        // P fragment (16x32 A layout)
        v16bf pf;
        {
            const __bf16* p = &Ps[wave][r][half * 8];
            pf = vcat(*(const v8bf*)p, *(const v8bf*)(p + 16));
        }
        // O += P @ V  (four 16x16 d tiles)
#pragma unroll
        for (int dc = 0; dc < 4; dc++) {
            const __bf16* p = &Vts[dc * 16 + r][half * 8];
            v16bf vf = vcat(*(const v8bf*)p, *(const v8bf*)(p + 16));
            o[dc] = wmma_bf16(pf, vf, o[dc]);
        }
    }
    __syncthreads();
    float linv[8];
#pragma unroll
    for (int j = 0; j < 8; j++) linv[j] = 1.f / lrow[wave][half * 8 + j];
    long long yb = (long long)b * Tv * Cv;
#pragma unroll
    for (int dc = 0; dc < 4; dc++)
#pragma unroll
        for (int j = 0; j < 8; j++) {
            int t = qb + half * 8 + j;
            Y[yb + (long long)t * Cv + h * Dv + dc * 16 + r] =
                (__bf16)(o[dc][j] * linv[j]);
        }
}

// ---------------------------------------------------------------------------
// Router: one wave per token; logits = z @ Wr + br; top-2 + softmax; counts
// ---------------------------------------------------------------------------
__global__ __launch_bounds__(256)
void router_kernel(const __bf16* __restrict__ z, const float* __restrict__ Wr,
                   const float* __restrict__ br, int* __restrict__ idx,
                   float* __restrict__ wgt, int* __restrict__ counts) {
    int tok  = blockIdx.x * (blockDim.x >> 5) + (threadIdx.x >> 5);
    int lane = threadIdx.x & 31;
    if (tok >= Nv) return;
    const __bf16* zr = z + (long long)tok * Cv;
    float acc[Ev] = {};
    for (int c = lane; c < Cv; c += 32) {
        float zv = (float)zr[c];
#pragma unroll
        for (int e = 0; e < Ev; e++) acc[e] += zv * Wr[c * Ev + e];
    }
#pragma unroll
    for (int e = 0; e < Ev; e++)
#pragma unroll
        for (int off = 16; off; off >>= 1) acc[e] += __shfl_xor(acc[e], off, 32);
    if (lane == 0) {
        float b0 = -1e30f, b1 = -1e30f;
        int i0 = 0, i1 = 0;
#pragma unroll
        for (int e = 0; e < Ev; e++) {
            float l = acc[e] + br[e];
            if (l > b0)      { b1 = b0; i1 = i0; b0 = l; i0 = e; }
            else if (l > b1) { b1 = l; i1 = e; }
        }
        float e0 = 1.f, e1 = __expf(b1 - b0), s = e0 + e1;
        idx[2 * tok] = i0;     idx[2 * tok + 1] = i1;
        wgt[2 * tok] = e0 / s; wgt[2 * tok + 1] = e1 / s;
        atomicAdd(&counts[i0], 1);
        atomicAdd(&counts[i1], 1);
    }
}

__global__ void zero_ints_kernel(int* p, int n) {
    int i = blockIdx.x * blockDim.x + threadIdx.x;
    if (i < n) p[i] = 0;
}

__global__ void moe_offsets_kernel(const int* __restrict__ counts,
                                   int* __restrict__ offs,
                                   int* __restrict__ cursor) {
    if (threadIdx.x == 0 && blockIdx.x == 0) {
        int s = 0;
        for (int e = 0; e < Ev; e++) {
            offs[e] = s; cursor[e] = s; s += counts[e];
        }
        offs[Ev] = s;
    }
}

__global__ void moe_scatter_kernel(const int* __restrict__ idx,
                                   const float* __restrict__ wgt,
                                   int* __restrict__ cursor,
                                   int* __restrict__ rowlist,
                                   float* __restrict__ wlist) {
    int t = blockIdx.x * blockDim.x + threadIdx.x;
    if (t >= Nv) return;
    for (int k = 0; k < 2; k++) {
        int e = idx[2 * t + k];
        int pos = atomicAdd(&cursor[e], 1);
        rowlist[pos] = t;
        wlist[pos]   = wgt[2 * t + k];
    }
}

// ---------------------------------------------------------------------------
// launch
// ---------------------------------------------------------------------------
extern "C" void kernel_launch(void* const* d_in, const int* in_sizes, int n_in,
                              void* d_out, int out_size, void* d_ws,
                              size_t ws_size, hipStream_t stream) {
    (void)in_sizes; (void)n_in; (void)out_size; (void)ws_size;
    const float* x     = (const float*)d_in[0];
    const float* ln1_g = (const float*)d_in[1];
    const float* ln1_b = (const float*)d_in[2];
    const float* Wqkv  = (const float*)d_in[3];
    const float* bqkv  = (const float*)d_in[4];
    const float* Wproj = (const float*)d_in[5];
    const float* bproj = (const float*)d_in[6];
    const float* ln2_g = (const float*)d_in[7];
    const float* ln2_b = (const float*)d_in[8];
    const float* Wr    = (const float*)d_in[9];
    const float* br    = (const float*)d_in[10];
    const float* W1    = (const float*)d_in[11];
    const float* b1    = (const float*)d_in[12];
    const float* W2    = (const float*)d_in[13];
    const float* b2    = (const float*)d_in[14];
    float* out = (float*)d_out;

    // workspace carve (256B aligned)
    char* w = (char*)d_ws;
    auto alloc = [&](size_t bytes) {
        void* p = (void*)w;
        w += (bytes + 255) & ~(size_t)255;
        return p;
    };
    __bf16* hb      = (__bf16*)alloc((size_t)Nv * Cv * 2);
    __bf16* Wqkv_b  = (__bf16*)alloc((size_t)Cv * 3 * Cv * 2);
    __bf16* Wproj_b = (__bf16*)alloc((size_t)Cv * Cv * 2);
    __bf16* W1_b    = (__bf16*)alloc((size_t)Ev * Cv * DFv * 2);
    __bf16* W2_b    = (__bf16*)alloc((size_t)Ev * DFv * Cv * 2);
    __bf16* qkv_b   = (__bf16*)alloc((size_t)Nv * 3 * Cv * 2);
    __bf16* qh      = (__bf16*)alloc((size_t)Nv * Cv * 2);
    __bf16* kh      = (__bf16*)alloc((size_t)Nv * Cv * 2);
    __bf16* vh      = (__bf16*)alloc((size_t)Nv * Cv * 2);
    __bf16* yb      = (__bf16*)alloc((size_t)Nv * Cv * 2);
    __bf16* zb      = (__bf16*)alloc((size_t)Nv * Cv * 2);
    int*    t_idx   = (int*)alloc((size_t)Nv * 2 * 4);
    float*  t_w     = (float*)alloc((size_t)Nv * 2 * 4);
    int*    counts  = (int*)alloc(Ev * 4);
    int*    offs    = (int*)alloc((Ev + 1) * 4);
    int*    cursor  = (int*)alloc(Ev * 4);
    int*    rowlist = (int*)alloc((size_t)Nv * 2 * 4);
    float*  wlist   = (float*)alloc((size_t)Nv * 2 * 4);
    __bf16* hid     = (__bf16*)alloc((size_t)2 * Nv * DFv * 2);

    // 1) weight casts to bf16
    cast_bf16_kernel<<<4096, 256, 0, stream>>>(Wqkv, Wqkv_b, (long long)Cv * 3 * Cv);
    cast_bf16_kernel<<<4096, 256, 0, stream>>>(Wproj, Wproj_b, (long long)Cv * Cv);
    cast_bf16_kernel<<<4096, 256, 0, stream>>>(W1, W1_b, (long long)Ev * Cv * DFv);
    cast_bf16_kernel<<<4096, 256, 0, stream>>>(W2, W2_b, (long long)Ev * DFv * Cv);

    // 2) LN1 -> h (bf16)
    ln_kernel<<<Nv / 8, 256, 0, stream>>>(x, ln1_g, ln1_b, hb, Nv, Cv);

    // 3) qkv = h @ Wqkv + bqkv (bf16 out)
    gemm_bf16<true, false, false, true, false, false>
        <<<dim3(3 * Cv / 128, Nv / 128, 1), 256, 0, stream>>>(
            hb, Cv, Wqkv_b, 0, bqkv, 0, nullptr, qkv_b, 3 * Cv,
            Nv, 3 * Cv, Cv, nullptr, nullptr, nullptr);

    // 4) split into per-head q/k/v
    split_qkv_kernel<<<4096, 256, 0, stream>>>(qkv_b, qh, kh, vh);

    // 5) flash attention -> y (bf16, [B,T,C])
    attn_kernel<<<dim3(Tv / 64, Hv, Bv), 128, 0, stream>>>(qh, kh, vh, yb);

    // 6) out = x + y @ Wproj + bproj (f32)
    gemm_bf16<true, true, false, false, false, false>
        <<<dim3(Cv / 128, Nv / 128, 1), 256, 0, stream>>>(
            yb, Cv, Wproj_b, 0, bproj, 0, x, out, Cv,
            Nv, Cv, Cv, nullptr, nullptr, nullptr);

    // 7) LN2 -> z (bf16)
    ln_kernel<<<Nv / 8, 256, 0, stream>>>(out, ln2_g, ln2_b, zb, Nv, Cv);

    // 8) router + top-2
    zero_ints_kernel<<<1, 64, 0, stream>>>(counts, Ev);
    router_kernel<<<Nv / 8, 256, 0, stream>>>(zb, Wr, br, t_idx, t_w, counts);
    moe_offsets_kernel<<<1, 32, 0, stream>>>(counts, offs, cursor);
    moe_scatter_kernel<<<Nv / 256, 256, 0, stream>>>(t_idx, t_w, cursor,
                                                     rowlist, wlist);

    // 9) hid = relu(z[gather] @ W1_e + b1_e) (bf16), grouped per expert
    gemm_bf16<true, false, true, true, true, false>
        <<<dim3(DFv / 128, (2 * Nv) / 128, Ev), 256, 0, stream>>>(
            zb, Cv, W1_b, (long long)Cv * DFv, b1, DFv, nullptr, hid, DFv,
            2 * Nv, DFv, Cv, rowlist, nullptr, offs);

    // 10) out += w * (hid @ W2_e + b2_e), weighted atomic scatter-add
    gemm_bf16<true, false, false, false, false, true>
        <<<dim3(Cv / 128, (2 * Nv) / 128, Ev), 256, 0, stream>>>(
            hid, DFv, W2_b, (long long)DFv * Cv, b2, Cv, nullptr, out, Cv,
            2 * Nv, Cv, DFv, rowlist, wlist, offs);
}
